// GraphAttentionEmbedding_62921270886788
// MI455X (gfx1250) — matrix operand order repaired and checked by reading, not compiled, not run on hardware
//
#include <hip/hip_runtime.h>
#include <hip/hip_fp16.h>

typedef __attribute__((ext_vector_type(16))) _Float16     v16h;
typedef __attribute__((ext_vector_type(8)))  float        v8f;
typedef __attribute__((ext_vector_type(4)))  unsigned int u32x4;
typedef __attribute__((ext_vector_type(2)))  unsigned int u32x2;

#define DD  128   // node feature dim
#define EDK 96    // edge attr dim (32 time enc + 64 msg)

// ---------------------------------------------------------------------------
// Load helpers (temporal-hint control for streaming vs reused data)
// ---------------------------------------------------------------------------
template <bool NT>
__device__ __forceinline__ u32x4 ld16B(const _Float16* p) {
  if (NT) return __builtin_nontemporal_load((const u32x4*)p);
  return *(const u32x4*)p;
}
template <bool NT>
__device__ __forceinline__ u32x2 ld8B(const _Float16* p) {
  if (NT) return __builtin_nontemporal_load((const u32x2*)p);
  return *(const u32x2*)p;
}

// ---------------------------------------------------------------------------
// Prep kernels
// ---------------------------------------------------------------------------
__global__ void f32_to_f16_kernel(const float* __restrict__ in,
                                  _Float16* __restrict__ out, long n) {
  long i = (long)blockIdx.x * blockDim.x + threadIdx.x;
  if (i < n) out[i] = (_Float16)in[i];
}

// W [K x 128] f32 row-major  ->  WT [128 x K] f16 (column-of-W contiguous)
__global__ void transpose_w_kernel(const float* __restrict__ W,
                                   _Float16* __restrict__ WT, int K) {
  int i = blockIdx.x * blockDim.x + threadIdx.x;
  if (i >= 128 * K) return;
  int n = i / K, k = i - n * K;
  WT[n * K + k] = (_Float16)W[k * 128 + n];
}

// edge_attr f16 [E x 96]: [cos(rel_t*tw+tb) (32) | msg (64)]  (streamed out NT)
__global__ void edge_attr_kernel(const int* __restrict__ src,
                                 const float* __restrict__ last_update,
                                 const float* __restrict__ t,
                                 const float* __restrict__ msg,
                                 const float* __restrict__ tw,
                                 const float* __restrict__ tb,
                                 _Float16* __restrict__ ea, int E) {
  long i = (long)blockIdx.x * blockDim.x + threadIdx.x;
  if (i >= (long)E * EDK) return;
  int e = (int)(i / EDK);
  int j = (int)(i - (long)e * EDK);
  float val;
  if (j < 32) {
    float rel = last_update[src[e]] - t[e];
    val = cosf(rel * tw[j] + tb[j]);
  } else {
    val = __builtin_nontemporal_load(&msg[(long)e * 64 + (j - 32)]);
  }
  __builtin_nontemporal_store((_Float16)val, &ea[i]);
}

// ---------------------------------------------------------------------------
// WMMA GEMM:  Y[M x 128] = A[M x K]_f16 @ W[K x 128] (+ bias)
//   WT is W transposed: [128 x K] f16, so each output column is K contiguous.
//   Block = 128 threads = 4 waves; wave w owns columns [32w, 32w+32) as two
//   16x16 tiles. B fragments for ALL k-steps are loaded ONCE into registers,
//   then the block iterates MT M-tiles, streaming only A (NT for edge data).
// Fragment layouts per CDNA5 ISA (CDNA5 ISA):
//   A 16x32 f16: lane<16 holds row(lane) K {0..7,16..23}; lane>=16 K {8..15,24..31}
//   B 32x16 f16: lane holds col(lane&15), K {0..7,16..23} (lane<16) / {8..15,24..31} (lane>=16)
//   C/D 16x16 f32: VGPR r -> row r (lane<16) / row r+8 (lane>=16), col = lane&15
// ---------------------------------------------------------------------------
union AF { v16h v; u32x4 q[2]; };

template <int K, bool HALF_OUT, bool NTA, int MT>
__global__ void wmma_gemm_kernel(const _Float16* __restrict__ A,
                                 const _Float16* __restrict__ WT,
                                 const float* __restrict__ bias,
                                 void* __restrict__ Y, long M) {
  constexpr int KS = K / 32;
  const int lane = threadIdx.x & 31;
  const int wave = threadIdx.x >> 5;
  const int l15  = lane & 15;
  const int hi   = (lane & 16) ? 1 : 0;

  const int n0 = wave * 32 + l15;
  const int n1 = n0 + 16;
  const _Float16* pB0 = WT + (long)n0 * K;
  const _Float16* pB1 = WT + (long)n1 * K;

  // B resident in registers for the whole block lifetime
  AF b[KS][2];
#pragma unroll
  for (int ks = 0; ks < KS; ++ks) {
    const int kb = ks * 32 + (hi ? 8 : 0);
    b[ks][0].q[0] = *(const u32x4*)(pB0 + kb);
    b[ks][0].q[1] = *(const u32x4*)(pB0 + kb + 16);
    b[ks][1].q[0] = *(const u32x4*)(pB1 + kb);
    b[ks][1].q[1] = *(const u32x4*)(pB1 + kb + 16);
  }
  const float bias0 = bias ? bias[n0] : 0.0f;
  const float bias1 = bias ? bias[n1] : 0.0f;

  const long tile0 = (long)blockIdx.x * MT;
#pragma unroll 1
  for (int mt = 0; mt < MT; ++mt) {
    const long mBase = (tile0 + mt) * 16;
    if (mBase >= M) return;
    long rowA = mBase + l15;
    if (rowA >= M) rowA = M - 1;         // clamp (keeps 16B alignment)
    const _Float16* pA = A + rowA * K;

    v8f acc0 = {};
    v8f acc1 = {};
#pragma unroll
    for (int ks = 0; ks < KS; ++ks) {
      AF a;
      const int ka = ks * 32 + (hi ? 8 : 0);
      a.q[0] = ld16B<NTA>(pA + ka);
      a.q[1] = ld16B<NTA>(pA + ka + 16);
      acc0 = __builtin_amdgcn_wmma_f32_16x16x32_f16(false, a.v, false, b[ks][0].v,
                                                    (short)0, acc0, false, false);
      acc1 = __builtin_amdgcn_wmma_f32_16x16x32_f16(false, a.v, false, b[ks][1].v,
                                                    (short)0, acc1, false, false);
    }

    const long mrow = mBase + (hi ? 8 : 0);
#pragma unroll
    for (int r = 0; r < 8; ++r) {
      const long m = mrow + r;
      if (m >= M) continue;
      if (HALF_OUT) {
        _Float16* O = (_Float16*)Y;
        __builtin_nontemporal_store((_Float16)(acc0[r] + bias0), &O[m * DD + n0]);
        __builtin_nontemporal_store((_Float16)(acc1[r] + bias1), &O[m * DD + n1]);
      } else {
        float* O = (float*)Y;
        O[m * DD + n0] = acc0[r] + bias0;
        O[m * DD + n1] = acc1[r] + bias1;
      }
    }
  }
}

// ---------------------------------------------------------------------------
// Attention passes (wave per edge; lane owns 4 channels; head = lane>>4)
// ---------------------------------------------------------------------------
__device__ __forceinline__ unsigned fenc(float f) {
  unsigned u = __float_as_uint(f);
  return (u & 0x80000000u) ? ~u : (u | 0x80000000u);
}
__device__ __forceinline__ float fdec(unsigned u) {
  return __uint_as_float((u & 0x80000000u) ? (u & 0x7FFFFFFFu) : ~u);
}

// Pass 1: alpha[e][h] = dot(q[dst], k[src]+e) / 8 ; segment max via atomicMax
__global__ void attn_alpha_kernel(const int* __restrict__ srcA,
                                  const int* __restrict__ dstA,
                                  const float* __restrict__ q,
                                  const float* __restrict__ k,
                                  const _Float16* __restrict__ ebuf,
                                  float* __restrict__ alpha,
                                  unsigned* __restrict__ amax, int E) {
  const int gid  = blockIdx.x * blockDim.x + threadIdx.x;
  const int e    = gid >> 5;
  const int lane = gid & 31;
  if (e >= E) return;
  const int s = srcA[e], d = dstA[e];
  const int c0 = lane * 4;
  const float4 q4 = *(const float4*)(q + (long)d * DD + c0);   // reused: RT
  const float4 k4 = *(const float4*)(k + (long)s * DD + c0);   // reused: RT
  union { u32x2 u; _Float16 h[4]; } e4;
  e4.u = ld8B<true>(ebuf + (long)e * DD + c0);                 // streamed: NT
  float p = q4.x * (k4.x + (float)e4.h[0]) + q4.y * (k4.y + (float)e4.h[1]) +
            q4.z * (k4.z + (float)e4.h[2]) + q4.w * (k4.w + (float)e4.h[3]);
  // reduce within each 16-lane half-wave (one head per half-wave)
  p += __shfl_xor(p, 1);
  p += __shfl_xor(p, 2);
  p += __shfl_xor(p, 4);
  p += __shfl_xor(p, 8);
  if ((lane & 15) == 0) {
    const int h = lane >> 4;
    const float al = p * 0.125f;  // / sqrt(C=64)
    alpha[(long)e * 2 + h] = al;
    atomicMax(&amax[(long)d * 2 + h], fenc(al));
  }
}

// Pass 2: ex = exp(alpha - amax[dst]); den += ex; num[dst] += ex*(v[src]+e)
__global__ void attn_aggr_kernel(const int* __restrict__ srcA,
                                 const int* __restrict__ dstA,
                                 const float* __restrict__ v,
                                 const _Float16* __restrict__ ebuf,
                                 const float* __restrict__ alpha,
                                 const unsigned* __restrict__ amax,
                                 float* __restrict__ den,
                                 float* __restrict__ num, int E) {
  const int gid  = blockIdx.x * blockDim.x + threadIdx.x;
  const int e    = gid >> 5;
  const int lane = gid & 31;
  if (e >= E) return;
  const int s = srcA[e], d = dstA[e];
  const int h = lane >> 4;
  const float ex = expf(alpha[(long)e * 2 + h] - fdec(amax[(long)d * 2 + h]));
  if ((lane & 15) == 0) atomicAdd(&den[(long)d * 2 + h], ex);
  const int c0 = lane * 4;
  const float4 v4 = *(const float4*)(v + (long)s * DD + c0);   // reused: RT
  union { u32x2 u; _Float16 hh[4]; } e4;
  e4.u = ld8B<true>(ebuf + (long)e * DD + c0);                 // streamed: NT
  float* o = num + (long)d * DD + c0;
  atomicAdd(o + 0, ex * (v4.x + (float)e4.hh[0]));
  atomicAdd(o + 1, ex * (v4.y + (float)e4.hh[1]));
  atomicAdd(o + 2, ex * (v4.z + (float)e4.hh[2]));
  atomicAdd(o + 3, ex * (v4.w + (float)e4.hh[3]));
}

// Layer-1 finalize: h1 = num/den + skip ; accumulate BN column sums
__global__ void finalize_bnstats_kernel(const float* __restrict__ num,
                                        const float* __restrict__ den,
                                        const float* __restrict__ skip,
                                        float* __restrict__ h1,
                                        float* __restrict__ colsum,
                                        float* __restrict__ colsq,
                                        int N, int rowsPerBlock) {
  const int c  = threadIdx.x;  // 0..127
  const int h  = c >> 6;
  int r0 = blockIdx.x * rowsPerBlock;
  int r1 = r0 + rowsPerBlock; if (r1 > N) r1 = N;
  float s = 0.f, s2 = 0.f;
  for (int r = r0; r < r1; ++r) {
    const float dn  = den[(long)r * 2 + h];
    const float agg = (dn > 0.f) ? num[(long)r * DD + c] / dn : 0.f;
    const float val = agg + skip[(long)r * DD + c];
    h1[(long)r * DD + c] = val;
    s += val; s2 += val * val;
  }
  atomicAdd(&colsum[c], s);
  atomicAdd(&colsq[c], s2);
}

// BN (biased var) + ReLU, emit f16 for layer-2 GEMMs
__global__ void bn_relu_f16_kernel(const float* __restrict__ h1,
                                   const float* __restrict__ colsum,
                                   const float* __restrict__ colsq,
                                   const float* __restrict__ g,
                                   const float* __restrict__ b,
                                   _Float16* __restrict__ out, int N) {
  long i = (long)blockIdx.x * blockDim.x + threadIdx.x;
  if (i >= (long)N * DD) return;
  const int c = (int)(i & (DD - 1));
  const float invN = 1.0f / (float)N;
  const float mu  = colsum[c] * invN;
  const float var = colsq[c] * invN - mu * mu;
  float val = (h1[i] - mu) * rsqrtf(var + 1e-5f) * g[c] + b[c];
  out[i] = (_Float16)(val > 0.f ? val : 0.f);
}

// Layer-2 finalize: out = num/den + skip
__global__ void finalize_out_kernel(const float* __restrict__ num,
                                    const float* __restrict__ den,
                                    const float* __restrict__ skip,
                                    float* __restrict__ out, int N) {
  long i = (long)blockIdx.x * blockDim.x + threadIdx.x;
  if (i >= (long)N * DD) return;
  const int c = (int)(i & (DD - 1));
  const long r = i >> 7;
  const float dn = den[r * 2 + (c >> 6)];
  out[i] = ((dn > 0.f) ? num[i] / dn : 0.f) + skip[i];
}

// ---------------------------------------------------------------------------
// Host launcher
// ---------------------------------------------------------------------------
extern "C" void kernel_launch(void* const* d_in, const int* in_sizes, int n_in,
                              void* d_out, int out_size, void* d_ws, size_t ws_size,
                              hipStream_t stream) {
  (void)n_in; (void)out_size; (void)ws_size;
  const float* x   = (const float*)d_in[0];
  const float* lu  = (const float*)d_in[1];
  const int*   eix = (const int*)d_in[2];
  const float* t   = (const float*)d_in[3];
  const float* msg = (const float*)d_in[4];
  const float* tw  = (const float*)d_in[5];
  const float* tb  = (const float*)d_in[6];
  const float* bng = (const float*)d_in[7];
  const float* bnb = (const float*)d_in[8];
  // layer weights: base 9 + 9*l : Wq,bq,Wk,bk,Wv,bv,We,Ws,bs
  const float *Wq[2], *bq[2], *Wk[2], *bk[2], *Wv[2], *bv[2], *We[2], *Ws[2], *bs[2];
  for (int l = 0; l < 2; ++l) {
    const int o = 9 + 9 * l;
    Wq[l] = (const float*)d_in[o + 0]; bq[l] = (const float*)d_in[o + 1];
    Wk[l] = (const float*)d_in[o + 2]; bk[l] = (const float*)d_in[o + 3];
    Wv[l] = (const float*)d_in[o + 4]; bv[l] = (const float*)d_in[o + 5];
    We[l] = (const float*)d_in[o + 6];
    Ws[l] = (const float*)d_in[o + 7]; bs[l] = (const float*)d_in[o + 8];
  }
  const int N = in_sizes[1];   // last_update is (N,)
  const int E = in_sizes[3];   // t is (E,)
  const int* src = eix;
  const int* dst = eix + E;

  // workspace carve-out
  char* p = (char*)d_ws;
  auto carve = [&](size_t bytes) {
    char* r = p;
    p += (bytes + 255) & ~(size_t)255;
    return r;
  };
  _Float16* xf16  = (_Float16*)carve((size_t)N * DD * 2);
  _Float16* hf16  = (_Float16*)carve((size_t)N * DD * 2);
  _Float16* ea16  = (_Float16*)carve((size_t)E * EDK * 2);
  _Float16* e16   = (_Float16*)carve((size_t)E * DD * 2);
  float*    qbuf  = (float*)carve((size_t)N * DD * 4);
  float*    kbuf  = (float*)carve((size_t)N * DD * 4);
  float*    vbuf  = (float*)carve((size_t)N * DD * 4);
  float*    sbuf  = (float*)carve((size_t)N * DD * 4);
  float*    num   = (float*)carve((size_t)N * DD * 4);
  float*    h1    = (float*)carve((size_t)N * DD * 4);
  float*    alpha = (float*)carve((size_t)E * 2 * 4);
  unsigned* amax  = (unsigned*)carve((size_t)N * 2 * 4);
  float*    den   = (float*)carve((size_t)N * 2 * 4);
  float*    colsum= (float*)carve(DD * 4);
  float*    colsq = (float*)carve(DD * 4);
  _Float16* wqT[2]; _Float16* wkT[2]; _Float16* wvT[2]; _Float16* wsT[2]; _Float16* weT[2];
  for (int l = 0; l < 2; ++l) {
    wqT[l] = (_Float16*)carve(128 * 128 * 2);
    wkT[l] = (_Float16*)carve(128 * 128 * 2);
    wvT[l] = (_Float16*)carve(128 * 128 * 2);
    wsT[l] = (_Float16*)carve(128 * 128 * 2);
    weT[l] = (_Float16*)carve(128 * EDK * 2);
  }

  const int TPB = 256;
  // ---- prep ----
  {
    long n = (long)N * DD;
    f32_to_f16_kernel<<<(unsigned)((n + TPB - 1) / TPB), TPB, 0, stream>>>(x, xf16, n);
  }
  for (int l = 0; l < 2; ++l) {
    const int nw = 128 * 128, ne = 128 * EDK;
    transpose_w_kernel<<<(nw + TPB - 1) / TPB, TPB, 0, stream>>>(Wq[l], wqT[l], 128);
    transpose_w_kernel<<<(nw + TPB - 1) / TPB, TPB, 0, stream>>>(Wk[l], wkT[l], 128);
    transpose_w_kernel<<<(nw + TPB - 1) / TPB, TPB, 0, stream>>>(Wv[l], wvT[l], 128);
    transpose_w_kernel<<<(nw + TPB - 1) / TPB, TPB, 0, stream>>>(Ws[l], wsT[l], 128);
    transpose_w_kernel<<<(ne + TPB - 1) / TPB, TPB, 0, stream>>>(We[l], weT[l], EDK);
  }
  {
    long n = (long)E * EDK;
    edge_attr_kernel<<<(unsigned)((n + TPB - 1) / TPB), TPB, 0, stream>>>(
        src, lu, t, msg, tw, tb, ea16, E);
  }

  const long tilesN = (N + 15) / 16;
  const long tilesE = (E + 15) / 16;
  const unsigned gridNg = (unsigned)((tilesN + 1) / 2);   // MT=2 for node GEMMs
  const unsigned gridEg = (unsigned)((tilesE + 7) / 8);   // MT=8 for edge GEMM
  const unsigned gridEw = (unsigned)(((long)E * 32 + TPB - 1) / TPB);
  const unsigned gridNC = (unsigned)(((long)N * DD + TPB - 1) / TPB);

  for (int l = 0; l < 2; ++l) {
    const _Float16* hin = (l == 0) ? xf16 : hf16;
    // e = edge_attr @ We  (WMMA, f16 out, NT-streamed A, B register-resident)
    wmma_gemm_kernel<EDK, true, true, 8><<<gridEg, 128, 0, stream>>>(
        ea16, weT[l], nullptr, e16, (long)E);
    // q,k,v,skip node GEMMs (WMMA, f32 out)
    wmma_gemm_kernel<DD, false, false, 2><<<gridNg, 128, 0, stream>>>(
        hin, wqT[l], bq[l], qbuf, (long)N);
    wmma_gemm_kernel<DD, false, false, 2><<<gridNg, 128, 0, stream>>>(
        hin, wkT[l], bk[l], kbuf, (long)N);
    wmma_gemm_kernel<DD, false, false, 2><<<gridNg, 128, 0, stream>>>(
        hin, wvT[l], bv[l], vbuf, (long)N);
    wmma_gemm_kernel<DD, false, false, 2><<<gridNg, 128, 0, stream>>>(
        hin, wsT[l], bs[l], sbuf, (long)N);
    // reset segment buffers
    (void)hipMemsetAsync(amax, 0, (size_t)N * 2 * 4, stream);
    (void)hipMemsetAsync(den,  0, (size_t)N * 2 * 4, stream);
    (void)hipMemsetAsync(num,  0, (size_t)N * DD * 4, stream);
    if (l == 0) {
      (void)hipMemsetAsync(colsum, 0, DD * 4, stream);
      (void)hipMemsetAsync(colsq,  0, DD * 4, stream);
    }
    // attention
    attn_alpha_kernel<<<gridEw, TPB, 0, stream>>>(src, dst, qbuf, kbuf, e16, alpha, amax, E);
    attn_aggr_kernel<<<gridEw, TPB, 0, stream>>>(src, dst, vbuf, e16, alpha, amax, den, num, E);
    if (l == 0) {
      const int rowsPerBlock = 512;
      finalize_bnstats_kernel<<<(N + rowsPerBlock - 1) / rowsPerBlock, 128, 0, stream>>>(
          num, den, sbuf, h1, colsum, colsq, N, rowsPerBlock);
      bn_relu_f16_kernel<<<gridNC, TPB, 0, stream>>>(h1, colsum, colsq, bng, bnb, hf16, N);
    } else {
      finalize_out_kernel<<<gridNC, TPB, 0, stream>>>(num, den, sbuf, (float*)d_out, N);
    }
  }
}
